// GaussianCrossingStatisticLayer_5085241278853
// MI455X (gfx1250) — compile-verified, matrix operand order checked by reading, not run on hardware
//
#include <hip/hip_runtime.h>

// GaussianCrossingStatistic on MI455X (gfx1250, wave32).
// out[e] = (1/128) * sum_t 1{ x[e] + noise[t,e] > 0.2 }
//
// Bandwidth-bound: 512 MiB noise streamed once (NT hints, no L2 pollution),
// ~22 us floor at 23.3 TB/s. The 128-way t-reduction is folded into ONE
// v_wmma_f32_16x16x128_fp8_fp8 per 16 outputs:
//   A = ones (fp8 E4M3 1.0 = 0x38), B = fp8 indicator matrix {0x00, 0x38},
//   D[m][n] = crossing count of column n, exact integers in f32.
// A is constant-ones => any bijective K(t) assignment inside B is correct;
// only the column<->lane (lane%16) mapping matters.
//
// Addressing: every noise load is written as (uniform row base) +
// (loop-invariant 32-bit per-lane offset) so the compiler emits GVS
// saddr-form global loads; row stepping becomes SALU, not 64-bit VALU adds.

typedef __attribute__((ext_vector_type(16))) int   v16i;
typedef __attribute__((ext_vector_type(8)))  float v8f;

#define NUM_T   128
#define HTHRESH 0.2f

__global__ __launch_bounds__(256) void gcs_wmma_kernel(
    const float* __restrict__ x,
    const float* __restrict__ noise,
    float* __restrict__ out,
    int total)                      // total = B*D elements (1048576)
{
    const int lane = threadIdx.x & 31;
    const int wave = blockIdx.x * (blockDim.x >> 5) + (threadIdx.x >> 5);
    const long base = (long)wave * 16;          // 16 output columns per wave
    if (base >= total) return;                  // wave-uniform exit only

    const int col  = lane & 15;                 // B/D column index (N = lane%16)
    const int half = lane >> 4;                 // t-half: [0,64) or [64,128)

    long elem = base + col;
    const bool valid = (elem < total);
    if (!valid) elem = (long)total - 1;         // clamp loads, keep EXEC all-1s for WMMA

    const float xv = __builtin_nontemporal_load(&x[elem]);

    // Per-lane, loop-invariant 32-bit byte offset (max ~272 MB, fits i32):
    // selects the lane's column and its t-half.
    const unsigned loff = (unsigned)half * (unsigned)(NUM_T / 2) * (unsigned)total * 4u
                        + (unsigned)elem * 4u;
    const char* nbytes = (const char*)noise;

    v16i Bm;                                    // 16 VGPRs x 4 fp8 bytes = 64 t-values/lane
#pragma unroll
    for (int v = 0; v < 16; ++v) {
        unsigned w = 0;
#pragma unroll
        for (int j = 0; j < 4; ++j) {
            const int r = v * 4 + j;            // uniform row 0..63 within the t-half
            const float* p = (const float*)(nbytes + (size_t)r * (size_t)total * 4u + loff);
            const float nv = __builtin_nontemporal_load(p);
            // keep the reference's exact numerics: (x + n) > H, then fp8 1.0 = 0x38
            w |= ((xv + nv > HTHRESH) ? 0x38u : 0u) << (8 * j);
        }
        Bm[v] = (int)w;
    }

    v16i Am;                                    // all-ones fp8 A matrix (16x128)
#pragma unroll
    for (int v = 0; v < 16; ++v) Am[v] = 0x38383838;

    v8f C = {};
    // D[m][n] = sum_{k=0..127} A[m][k]*B[k][n] = crossing count for column n
    C = __builtin_amdgcn_wmma_f32_16x16x128_fp8_fp8(Am, Bm, (short)0, C,
                                                    /*reuse_a=*/false,
                                                    /*reuse_b=*/false);

    // D layout: VGPR0, lanes 0-15 = row M=0, N=lane. Counts are exact integers.
    if (lane < 16 && valid) {
        __builtin_nontemporal_store(C[0] * (1.0f / NUM_T), &out[elem]);
    }
}

extern "C" void kernel_launch(void* const* d_in, const int* in_sizes, int n_in,
                              void* d_out, int out_size, void* d_ws, size_t ws_size,
                              hipStream_t stream) {
    const float* x     = (const float*)d_in[0];   // [32, 32768] f32
    const float* noise = (const float*)d_in[1];   // [128, 32, 32768] f32
    float*       out   = (float*)d_out;           // [32, 32768] f32

    const int total  = in_sizes[0];               // 1048576 elements
    const int waves  = (total + 15) / 16;         // 16 outputs per wave
    const int wavesPerBlock = 256 / 32;           // wave32: 8 waves/block
    const int blocks = (waves + wavesPerBlock - 1) / wavesPerBlock;

    gcs_wmma_kernel<<<blocks, 256, 0, stream>>>(x, noise, out, total);
}
